// TransformerBlock_81475529605501
// MI455X (gfx1250) — matrix operand order repaired and checked by reading, not compiled, not run on hardware
//
#include <hip/hip_runtime.h>

// ---------------------------------------------------------------------------
// Types for CDNA5 WMMA fragments (wave32)
// ---------------------------------------------------------------------------
typedef __attribute__((ext_vector_type(16))) __bf16 v16bf;
typedef __attribute__((ext_vector_type(8)))  __bf16 v8bf;
typedef __attribute__((ext_vector_type(8)))  float  v8f;

constexpr int Bsz = 16;
constexpr int Ssz = 512;
constexpr int Dsz = 768;
constexpr int Hh  = 12;
constexpr int DHd = 64;
constexpr int FFs = 3072;
constexpr int Mrows = Bsz * Ssz;     // 8192

// Epilogue modes for the generic GEMM
constexpr int EPI_QK   = 0;  // bf16 out remapped to [b,h,s,d]
constexpr int EPI_V    = 1;  // bf16 out remapped to [b,h,d,s]  (transposed)
constexpr int EPI_RES  = 2;  // f32 out = acc + bias + resid
constexpr int EPI_GELU = 3;  // bf16 out = gelu(acc + bias)

static __device__ __forceinline__ v8f wmma_bf(v16bf a, v16bf b, v8f c) {
  // D = A(16x32 bf16) * B(32x16 bf16) + C(16x16 f32)
  return __builtin_amdgcn_wmma_f32_16x16x32_bf16(
      /*neg_a=*/false, a, /*neg_b=*/false, b,
      /*c_mod=*/(short)0, c, /*reuse_a=*/false, /*reuse_b=*/false);
}

// A-fragment (16x32 bf16): lane = row (lane&15); K-halves at kb=(lane>>4)*16
// VGPR0..3 <- K[kb..kb+7], VGPR4..7 <- K[kb+8..kb+15]
static __device__ __forceinline__ v16bf load_a_frag(const __bf16* p) {
  union { v16bf v; v8bf h[2]; } u;
  u.h[0] = *(const v8bf*)(p);
  u.h[1] = *(const v8bf*)(p + 8);
  return u.v;
}

// Exchange a float with lane^16 (cross 16-lane halves of a wave32)
static __device__ __forceinline__ float xhalf(float v) {
#if __has_builtin(__builtin_amdgcn_permlanex16)
  int i = __builtin_bit_cast(int, v);
  i = __builtin_amdgcn_permlanex16(i, i, 0x76543210, 0xFEDCBA98,
                                   /*fi=*/false, /*bound_ctrl=*/false);
  return __builtin_bit_cast(float, i);
#else
  return __shfl_xor(v, 16, 32);
#endif
}

static __device__ __forceinline__ float gelu_f(float x) {
  float x3 = x * x * x;
  return 0.5f * x * (1.0f + tanhf(0.79788456080286536f * (x + 0.044715f * x3)));
}

// ---------------------------------------------------------------------------
// fp32 -> bf16 converts (weights pre-transposed so B-fragments are contiguous)
// ---------------------------------------------------------------------------
__global__ __launch_bounds__(256) void cvt_bf_k(const float* __restrict__ in,
                                                __bf16* __restrict__ out, int n) {
  int i = blockIdx.x * 256 + threadIdx.x;
  if (i < n) out[i] = (__bf16)in[i];
}

// in: [K][N] row-major (as stored by the reference, x @ W).  out: [N][K] = W^T
__global__ __launch_bounds__(256) void cvt_t_k(const float* __restrict__ in,
                                               __bf16* __restrict__ out,
                                               int K, int N) {
  int i = blockIdx.x * 256 + threadIdx.x;
  if (i < K * N) {
    int n = i / K;
    int k = i - n * K;
    out[i] = (__bf16)in[(size_t)k * N + n];
  }
}

// ---------------------------------------------------------------------------
// Generic bf16 WMMA GEMM: C[M,N] = A[M,K] * Bt[N,K]^T (+bias, epilogue)
// Block = 256 threads = 8 waves; block tile 128x128; wave tile 32x64.
// ---------------------------------------------------------------------------
template <int EPI>
__global__ __launch_bounds__(256)
void gemm_bf16_k(const __bf16* __restrict__ A, const __bf16* __restrict__ Bt,
                 const float* __restrict__ bias, const float* __restrict__ resid,
                 float* __restrict__ Cf, __bf16* __restrict__ Cb,
                 int M, int N, int K) {
  const int lane = threadIdx.x & 31;
  const int wave = threadIdx.x >> 5;
  const int lo = lane & 15;
  const int hi = lane >> 4;
  const int wm = wave & 3;          // 4 waves along M
  const int wn = wave >> 2;         // 2 waves along N
  const int mbase = blockIdx.y * 128 + wm * 32;
  const int nbase = blockIdx.x * 128 + wn * 64;

  const v8f vzero = {0.f, 0.f, 0.f, 0.f, 0.f, 0.f, 0.f, 0.f};
  v8f acc[2][4];
#pragma unroll
  for (int i = 0; i < 2; ++i)
#pragma unroll
    for (int j = 0; j < 4; ++j) acc[i][j] = vzero;

  const __bf16* arow0 = A + (size_t)(mbase + lo) * K + (hi << 4);
  const __bf16* arow1 = arow0 + (size_t)16 * K;
  const __bf16* brow[4];
#pragma unroll
  for (int j = 0; j < 4; ++j)
    brow[j] = Bt + (size_t)(nbase + j * 16 + lo) * K + (hi << 4);

  for (int k0 = 0; k0 < K; k0 += 32) {
    if (k0 + 64 < K) {               // global_prefetch_b8 hints for next tiles
      __builtin_prefetch(arow0 + k0 + 64, 0, 1);
      __builtin_prefetch(brow[0] + k0 + 64, 0, 1);
      __builtin_prefetch(brow[2] + k0 + 64, 0, 1);
    }
    v16bf a0 = load_a_frag(arow0 + k0);
    v16bf a1 = load_a_frag(arow1 + k0);
#pragma unroll
    for (int j = 0; j < 4; ++j) {
      v16bf b = *(const v16bf*)(brow[j] + k0);
      acc[0][j] = wmma_bf(a0, b, acc[0][j]);
      acc[1][j] = wmma_bf(a1, b, acc[1][j]);
    }
  }

  // Epilogue.  C layout: VGPR r, lanes 0-15 -> M=r, lanes 16-31 -> M=r+8; N=lane&15.
#pragma unroll
  for (int i = 0; i < 2; ++i) {
#pragma unroll
    for (int j = 0; j < 4; ++j) {
      const int n = nbase + j * 16 + lo;
      const float bv = bias[n];
#pragma unroll
      for (int r = 0; r < 8; ++r) {
        const int m = mbase + i * 16 + r + hi * 8;
        float val = acc[i][j][r] + bv;
        if constexpr (EPI == EPI_RES) {
          Cf[(size_t)m * N + n] = val + resid[(size_t)m * N + n];
        } else if constexpr (EPI == EPI_GELU) {
          Cb[(size_t)m * N + n] = (__bf16)gelu_f(val);
        } else if constexpr (EPI == EPI_QK) {
          const int b = m >> 9, s = m & 511, h = n >> 6, d = n & 63;
          Cb[(((size_t)(b * Hh + h)) * Ssz + s) * DHd + d] = (__bf16)val;
        } else {  // EPI_V: transposed per-head V layout [b,h,d,s]
          const int b = m >> 9, s = m & 511, h = n >> 6, d = n & 63;
          Cb[(((size_t)(b * Hh + h)) * DHd + d) * Ssz + s] = (__bf16)val;
        }
      }
    }
  }
}

// ---------------------------------------------------------------------------
// Flash attention, transposed-score formulation (no LDS, scalar row stats).
//   S^T(k,q) tiles = K(A-frag, natural [b,h,s,d]) x Q^T(B-frag, natural layout)
//   -> each lane holds 16 key-scores of ONE query; softmax needs only
//      VALU reductions + one permlanex16 per stat.
//   O^T(d,q) tiles = V^T(A-frag, [b,h,d,s]) x P^T(B-frag built in registers)
// One wave = 16 queries (q = qbase + lane&15).  Block = 8 waves = 128 queries.
// ---------------------------------------------------------------------------
__global__ __launch_bounds__(256)
void attn_k(const __bf16* __restrict__ Q, const __bf16* __restrict__ Km,
            const __bf16* __restrict__ Vt, const unsigned char* __restrict__ mask,
            __bf16* __restrict__ out) {
  const int lane = threadIdx.x & 31;
  const int wave = threadIdx.x >> 5;
  const int lo = lane & 15;
  const int hi = lane >> 4;
  const int bh = blockIdx.y;
  const int b = bh / Hh;
  const int h = bh - b * Hh;
  const int qbase = blockIdx.x * 128 + wave * 16;

  const __bf16* Qb = Q  + (size_t)bh * Ssz * DHd;
  const __bf16* Kb = Km + (size_t)bh * Ssz * DHd;
  const __bf16* Vb = Vt + (size_t)bh * DHd * Ssz;
  const unsigned char* maskrow = mask + b * Ssz;

  // Q^T B-fragments (d=0..31, d=32..63), hoisted: reused by every K tile
  const v16bf bq0 = *(const v16bf*)(Qb + (size_t)(qbase + lo) * DHd + (hi << 4));
  const v16bf bq1 = *(const v16bf*)(Qb + (size_t)(qbase + lo) * DHd + 32 + (hi << 4));

  const v8f vzero = {0.f, 0.f, 0.f, 0.f, 0.f, 0.f, 0.f, 0.f};
  v8f o[4] = {vzero, vzero, vzero, vzero};   // O^T tiles, d = t*16 + r + 8*hi
  float mrun = -1e30f;                        // per-lane (= per-query) stats
  float lrun = 0.f;

  for (int kc = 0; kc < Ssz; kc += 32) {
    // S^T tiles: k rows kc..kc+15 (s0) and kc+16..kc+31 (s1)
    v16bf ak00 = load_a_frag(Kb + (size_t)(kc + lo) * DHd + (hi << 4));
    v16bf ak01 = load_a_frag(Kb + (size_t)(kc + lo) * DHd + 32 + (hi << 4));
    v16bf ak10 = load_a_frag(Kb + (size_t)(kc + 16 + lo) * DHd + (hi << 4));
    v16bf ak11 = load_a_frag(Kb + (size_t)(kc + 16 + lo) * DHd + 32 + (hi << 4));
    v8f s0 = wmma_bf(ak00, bq0, vzero); s0 = wmma_bf(ak01, bq1, s0);
    v8f s1 = wmma_bf(ak10, bq0, vzero); s1 = wmma_bf(ak11, bq1, s1);

    // mask bytes for this lane's k positions: slot r of tile m -> k = kc + m*16 + r + 8*hi
    const unsigned long long mm0 =
        *(const unsigned long long*)(maskrow + kc + (hi << 3));
    const unsigned long long mm1 =
        *(const unsigned long long*)(maskrow + kc + 16 + (hi << 3));

    float xs0[8], xs1[8];
    float tmax = -1e30f;
#pragma unroll
    for (int r = 0; r < 8; ++r) {
      const float mv0 = ((mm0 >> (8 * r)) & 0xffULL) ? 1.f : 0.f;
      const float mv1 = ((mm1 >> (8 * r)) & 0xffULL) ? 1.f : 0.f;
      float x0 = s0[r] * 0.125f;  x0 = mv0 * (x0 + 100.f) - 100.f;
      float x1 = s1[r] * 0.125f;  x1 = mv1 * (x1 + 100.f) - 100.f;
      xs0[r] = x0; xs1[r] = x1;
      tmax = fmaxf(tmax, fmaxf(x0, x1));
    }
    tmax = fmaxf(tmax, xhalf(tmax));          // other 16 k's live in lane^16
    const float nm = fmaxf(mrun, tmax);
    const float corr = __expf(mrun - nm);
    mrun = nm;

    float p0[8], p1[8];
    float rs = 0.f;
#pragma unroll
    for (int r = 0; r < 8; ++r) {
      p0[r] = __expf(xs0[r] - nm);
      p1[r] = __expf(xs1[r] - nm);
      rs += p0[r] + p1[r];
    }
    rs += xhalf(rs);
    lrun = lrun * corr + rs;

#pragma unroll
    for (int t = 0; t < 4; ++t)
#pragma unroll
      for (int r = 0; r < 8; ++r) o[t][r] = o[t][r] * corr;

    // Build P^T B-fragment: lane needs k = kc + hi*16 + 0..15 for its query.
    // hi=0 owns k{0..7,16..23}, hi=1 owns k{8..15,24..31}: one cross-half swap.
    union { v16bf v; __bf16 e[16]; } bp;
#pragma unroll
    for (int r = 0; r < 8; ++r) {
      const float send = hi ? p0[r] : p1[r];
      const float recv = xhalf(send);
      const float lowv  = hi ? recv : p0[r];   // slots 0..7  = K base+0..7
      const float highv = hi ? p1[r] : recv;   // slots 8..15 = K base+8..15
      bp.e[r]     = (__bf16)lowv;
      bp.e[8 + r] = (__bf16)highv;
    }

    // O^T += V^T x P^T   (V^T A-frags contiguous from [b,h,d,s] layout)
#pragma unroll
    for (int t = 0; t < 4; ++t) {
      v16bf av = load_a_frag(Vb + (size_t)(t * 16 + lo) * Ssz + kc + (hi << 4));
      o[t] = wmma_bf(av, bp.v, o[t]);
    }
  }

  // normalize (scalar per lane) and store packed 16B rows of d per tile
  const float inv = 1.f / lrun;
  const size_t rowoff = ((size_t)(b * Ssz + qbase + lo)) * Dsz + h * DHd;
#pragma unroll
  for (int t = 0; t < 4; ++t) {
    union { v8bf v; __bf16 e[8]; } pk;
#pragma unroll
    for (int r = 0; r < 8; ++r) pk.e[r] = (__bf16)(o[t][r] * inv);
    *(v8bf*)(out + rowoff + t * 16 + 8 * hi) = pk.v;
  }
}

// ---------------------------------------------------------------------------
// LayerNorm over D=768, one row per block, 3 elements per thread
// ---------------------------------------------------------------------------
__global__ __launch_bounds__(256)
void ln_k(const float* __restrict__ y, const float* __restrict__ g,
          const float* __restrict__ be, float* __restrict__ outf,
          __bf16* __restrict__ outb) {
  const int row = blockIdx.x;
  const int tid = threadIdx.x;
  const float* yr = y + (size_t)row * Dsz;
  float v0 = yr[tid], v1 = yr[tid + 256], v2 = yr[tid + 512];

  __shared__ float red[256];
  red[tid] = v0 + v1 + v2;
  __syncthreads();
  for (int off = 128; off > 0; off >>= 1) {
    if (tid < off) red[tid] += red[tid + off];
    __syncthreads();
  }
  const float mean = red[0] * (1.f / 768.f);
  __syncthreads();

  float d0 = v0 - mean, d1 = v1 - mean, d2 = v2 - mean;
  red[tid] = d0 * d0 + d1 * d1 + d2 * d2;
  __syncthreads();
  for (int off = 128; off > 0; off >>= 1) {
    if (tid < off) red[tid] += red[tid + off];
    __syncthreads();
  }
  const float rstd = rsqrtf(red[0] * (1.f / 768.f) + 1e-11f);

  float dv[3] = {d0, d1, d2};
#pragma unroll
  for (int e = 0; e < 3; ++e) {
    const int idx = tid + e * 256;
    const float o = dv[e] * rstd * g[idx] + be[idx];
    if (outf) outf[(size_t)row * Dsz + idx] = o;
    if (outb) outb[(size_t)row * Dsz + idx] = (__bf16)o;
  }
}

// ---------------------------------------------------------------------------
// Launch
// ---------------------------------------------------------------------------
extern "C" void kernel_launch(void* const* d_in, const int* in_sizes, int n_in,
                              void* d_out, int out_size, void* d_ws, size_t ws_size,
                              hipStream_t stream) {
  (void)in_sizes; (void)n_in; (void)out_size; (void)ws_size;

  const float* x    = (const float*)d_in[0];
  const unsigned char* mask = (const unsigned char*)d_in[1];
  const float* Wq  = (const float*)d_in[2];   const float* bq  = (const float*)d_in[3];
  const float* Wk  = (const float*)d_in[4];   const float* bk  = (const float*)d_in[5];
  const float* Wv  = (const float*)d_in[6];   const float* bv  = (const float*)d_in[7];
  const float* Wo  = (const float*)d_in[8];   const float* bo  = (const float*)d_in[9];
  const float* g1  = (const float*)d_in[10];  const float* b1  = (const float*)d_in[11];
  const float* Wi  = (const float*)d_in[12];  const float* bi  = (const float*)d_in[13];
  const float* Wo2 = (const float*)d_in[14];  const float* bo2 = (const float*)d_in[15];
  const float* g2  = (const float*)d_in[16];  const float* b2  = (const float*)d_in[17];
  float* out = (float*)d_out;

  // workspace carve-up (aligned to 256B)
  char* wsp = (char*)d_ws;
  auto alloc = [&](size_t bytes) -> void* {
    void* p = (void*)wsp;
    wsp += (bytes + 255) & ~(size_t)255;
    return p;
  };
  const size_t nXD = (size_t)Mrows * Dsz;          // 8192*768
  const size_t nFF = (size_t)Mrows * FFs;          // 8192*3072
  __bf16* Xb     = (__bf16*)alloc(nXD * 2);
  __bf16* Wq_t   = (__bf16*)alloc((size_t)Dsz * Dsz * 2);
  __bf16* Wk_t   = (__bf16*)alloc((size_t)Dsz * Dsz * 2);
  __bf16* Wv_t   = (__bf16*)alloc((size_t)Dsz * Dsz * 2);
  __bf16* Wo_t   = (__bf16*)alloc((size_t)Dsz * Dsz * 2);
  __bf16* Wi_t   = (__bf16*)alloc((size_t)Dsz * FFs * 2);
  __bf16* Wo2_t  = (__bf16*)alloc((size_t)FFs * Dsz * 2);
  __bf16* Qh     = (__bf16*)alloc(nXD * 2);
  __bf16* Kh     = (__bf16*)alloc(nXD * 2);
  __bf16* Vt     = (__bf16*)alloc(nXD * 2);
  __bf16* attn_b = (__bf16*)alloc(nXD * 2);
  float*  y1     = (float*) alloc(nXD * 4);        // reused as y2
  float*  ln1f   = (float*) alloc(nXD * 4);
  __bf16* ln1b   = (__bf16*)alloc(nXD * 2);
  __bf16* interb = (__bf16*)alloc(nFF * 2);

  // 1) bf16 staging: x row-major, weights transposed [N][K]
  cvt_bf_k<<<(int)((nXD + 255) / 256), 256, 0, stream>>>(x, Xb, (int)nXD);
  const int nW = Dsz * Dsz, nWff = Dsz * FFs;
  cvt_t_k<<<(nW   + 255) / 256, 256, 0, stream>>>(Wq,  Wq_t,  Dsz, Dsz);
  cvt_t_k<<<(nW   + 255) / 256, 256, 0, stream>>>(Wk,  Wk_t,  Dsz, Dsz);
  cvt_t_k<<<(nW   + 255) / 256, 256, 0, stream>>>(Wv,  Wv_t,  Dsz, Dsz);
  cvt_t_k<<<(nW   + 255) / 256, 256, 0, stream>>>(Wo,  Wo_t,  Dsz, Dsz);
  cvt_t_k<<<(nWff + 255) / 256, 256, 0, stream>>>(Wi,  Wi_t,  Dsz, FFs);
  cvt_t_k<<<(nWff + 255) / 256, 256, 0, stream>>>(Wo2, Wo2_t, FFs, Dsz);

  // 2) QKV projections: Q,K natural per-head [b,h,s,d]; V transposed [b,h,d,s]
  dim3 gD(Dsz / 128, Mrows / 128);                  // (6, 64)
  gemm_bf16_k<EPI_QK><<<gD, 256, 0, stream>>>(Xb, Wq_t, bq, nullptr, nullptr, Qh, Mrows, Dsz, Dsz);
  gemm_bf16_k<EPI_QK><<<gD, 256, 0, stream>>>(Xb, Wk_t, bk, nullptr, nullptr, Kh, Mrows, Dsz, Dsz);
  gemm_bf16_k<EPI_V ><<<gD, 256, 0, stream>>>(Xb, Wv_t, bv, nullptr, nullptr, Vt, Mrows, Dsz, Dsz);

  // 3) flash attention: grid = (S/128 q-tiles, B*H)
  attn_k<<<dim3(Ssz / 128, Bsz * Hh), 256, 0, stream>>>(Qh, Kh, Vt, mask, attn_b);

  // 4) output projection + residual -> y1, then LN1
  gemm_bf16_k<EPI_RES><<<gD, 256, 0, stream>>>(attn_b, Wo_t, bo, x, y1, nullptr, Mrows, Dsz, Dsz);
  ln_k<<<Mrows, 256, 0, stream>>>(y1, g1, b1, ln1f, ln1b);

  // 5) FFN: gelu dense, then output dense + residual, then LN2 -> d_out
  gemm_bf16_k<EPI_GELU><<<dim3(FFs / 128, Mrows / 128), 256, 0, stream>>>(
      ln1b, Wi_t, bi, nullptr, nullptr, interb, Mrows, FFs, Dsz);
  gemm_bf16_k<EPI_RES><<<gD, 256, 0, stream>>>(interb, Wo2_t, bo2, ln1f, y1, nullptr, Mrows, Dsz, FFs);
  ln_k<<<Mrows, 256, 0, stream>>>(y1, g2, b2, out, nullptr);
}
